// Encoder_50852412785097
// MI455X (gfx1250) — matrix operand order repaired and checked by reading, not compiled, not run on hardware
//
#include <hip/hip_runtime.h>
#include <hip/hip_bf16.h>
#include <stdint.h>

// Problem constants (reference: B=2, S=2048, H=8, D=64)
constexpr int Bn = 2;
constexpr int Sn = 2048;
constexpr int Hn = 8;
constexpr int Dn = 64;
constexpr float SCALE = 0.125f;         // 1/sqrt(64)
constexpr float NEG_INF = -1e30f;

typedef __bf16 bf16_t;
typedef bf16_t v16bf __attribute__((ext_vector_type(16)));
typedef bf16_t v8bf  __attribute__((ext_vector_type(8)));
typedef float  v8f   __attribute__((ext_vector_type(8)));

__device__ __forceinline__ bf16_t f2bf(float f) { return (bf16_t)f; }  // native cvt
__device__ __forceinline__ bf16_t bits2bf(uint16_t u) {
  bf16_t o; __builtin_memcpy(&o, &u, 2); return o;
}

// A-fragment (16x32 bf16) element -> K index (contiguous order, matching B):
// lanes 0-15 hold K=0..15, lanes 16-31 hold K=16..31.
// Equivalently: element i -> K = half*16 + i.
__device__ __forceinline__ int a_k(int i, int half) {
  return half * 16 + i;
}

// Build A-frag from two contiguous 8-element runs (16B-aligned loads).
__device__ __forceinline__ v16bf load_afrag(const bf16_t* rowbase, int kt, int half) {
  const v8bf lo = *(const v8bf*)(rowbase + kt * 32 + half * 16);
  const v8bf hi = *(const v8bf*)(rowbase + kt * 32 + half * 16 + 8);
  return __builtin_shufflevector(lo, hi, 0, 1, 2, 3, 4, 5, 6, 7,
                                 8, 9, 10, 11, 12, 13, 14, 15);
}

__device__ __forceinline__ float redmax16(float v) {
#pragma unroll
  for (int off = 8; off; off >>= 1) v = fmaxf(v, __shfl_xor(v, off, 16));
  return v;
}
__device__ __forceinline__ float redsum16(float v) {
#pragma unroll
  for (int off = 8; off; off >>= 1) v += __shfl_xor(v, off, 16);
  return v;
}

// ---------------------------------------------------------------------------
// Prepass A: fp32 -> bf16 straight conversion (for X).
// ---------------------------------------------------------------------------
__global__ __launch_bounds__(256)
void enc_cvt_kernel(const float* __restrict__ in, bf16_t* __restrict__ out, int n) {
  int i = blockIdx.x * 256 + threadIdx.x;
  if (i < n) out[i] = f2bf(in[i]);
}

// ---------------------------------------------------------------------------
// Prepass B: batched [R x C] fp32 -> [C x R] bf16 transpose-convert
// (weights become N-major so B-fragments are contiguous v16bf loads).
// ---------------------------------------------------------------------------
__global__ __launch_bounds__(256)
void enc_trans_kernel(const float* __restrict__ in, bf16_t* __restrict__ out,
                      int R, int C) {
  const int mat = blockIdx.x;
  const int idx = blockIdx.y * 256 + threadIdx.x;
  if (idx >= R * C) return;
  const int r = idx / C, c = idx % C;
  out[(size_t)mat * R * C + (size_t)c * R + r] =
      f2bf(in[(size_t)mat * R * C + (size_t)r * C + c]);
}

// ---------------------------------------------------------------------------
// Kernel 1: per-head Q/K/V projection.  One wave computes a 16-row tile for
// one (b,h).  Q,K stored row-major [bh][s][d]; V stored transposed [bh][d][s]
// so the p-V B-fragment later is a contiguous v16bf load.
// ---------------------------------------------------------------------------
__global__ __launch_bounds__(32)
void enc_qkv_kernel(const bf16_t* __restrict__ Xb,
                    const bf16_t* __restrict__ wQt,
                    const bf16_t* __restrict__ wKt,
                    const bf16_t* __restrict__ wVt,
                    bf16_t* __restrict__ Qb,
                    bf16_t* __restrict__ Kb,
                    bf16_t* __restrict__ Vt) {
  const int l  = threadIdx.x;
  const int bh = blockIdx.x;
  const int b  = bh / Hn, h = bh % Hn;
  const int s0 = blockIdx.y * 16;
  const int m  = l & 15, half = l >> 4;
  const int n  = m;

  const bf16_t* xrow = Xb + ((size_t)b * Sn + s0 + m) * Dn;
  v16bf a[2];
#pragma unroll
  for (int kt = 0; kt < 2; ++kt) a[kt] = load_afrag(xrow, kt, half);

#pragma unroll
  for (int w = 0; w < 3; ++w) {
    const bf16_t* Wt = ((w == 0) ? wQt : (w == 1) ? wKt : wVt) +
                       (size_t)h * Dn * Dn;           // [e][k] per head
#pragma unroll
    for (int nt = 0; nt < 4; ++nt) {
      v8f c = {};
#pragma unroll
      for (int kt = 0; kt < 2; ++kt) {
        const v16bf bw =
            *(const v16bf*)(Wt + (size_t)(nt * 16 + n) * Dn + kt * 32 + half * 16);
        c = __builtin_amdgcn_wmma_f32_16x16x32_bf16(false, a[kt], false, bw,
                                                    (short)0, c, false, false);
      }
      if (w == 2) {   // V: transposed store [bh][e][s]
#pragma unroll
        for (int r = 0; r < 8; ++r)
          Vt[((size_t)bh * Dn + nt * 16 + n) * Sn + s0 + r + 8 * half] = f2bf(c[r]);
      } else {        // Q,K: row-major store [bh][s][d]
        bf16_t* Ot = (w == 0) ? Qb : Kb;
#pragma unroll
        for (int r = 0; r < 8; ++r)
          Ot[((size_t)bh * Sn + s0 + r + 8 * half) * Dn + nt * 16 + n] = f2bf(c[r]);
      }
    }
  }
}

// ---------------------------------------------------------------------------
// Kernel 2: fused two-phase attention (per 16-row tile, one wave):
//   phase 1: streaming QK^T -> first-softmax row max / exp-sum (online)
//   phase 2: streaming QK^T -> x, lane-group cumsum, time-decay reweight,
//            online second softmax, WMMA p-V accumulation.
// ---------------------------------------------------------------------------
__global__ __launch_bounds__(32)
void enc_attn_kernel(const bf16_t* __restrict__ Qb,
                     const bf16_t* __restrict__ Kb,
                     const bf16_t* __restrict__ Vt,
                     const int* __restrict__ tstamp,
                     const float* __restrict__ theta,
                     bf16_t* __restrict__ Ob) {
  __shared__ uint16_t ldsP[16 * 17];      // 16x16 prob tile, padded stride

  const int l  = threadIdx.x;
  const int bh = blockIdx.x;
  const int b  = bh / Hn;
  const int it = blockIdx.y;
  const int s0 = it * 16;
  const int m  = l & 15, half = l >> 4;
  const int n  = m;
  const size_t base = (size_t)bh * Sn;
  const float th  = theta[0];
  const float th2 = th * th;

  v16bf a[2];
  {
    const bf16_t* qrow = Qb + (base + s0 + m) * Dn;
#pragma unroll
    for (int kt = 0; kt < 2; ++kt) a[kt] = load_afrag(qrow, kt, half);
  }

  // ---- phase 1: first-softmax statistics ----
  float m1[8], d1[8];
#pragma unroll
  for (int r = 0; r < 8; ++r) { m1[r] = NEG_INF; d1[r] = 0.f; }

  for (int jt = 0; jt <= it; ++jt) {
    const int j0 = jt * 16;
    v8f c = {};
#pragma unroll
    for (int kt = 0; kt < 2; ++kt) {
      const v16bf bk =
          *(const v16bf*)(Kb + (base + j0 + n) * Dn + kt * 32 + half * 16);
      c = __builtin_amdgcn_wmma_f32_16x16x32_bf16(false, a[kt], false, bk,
                                                  (short)0, c, false, false);
    }
#pragma unroll
    for (int r = 0; r < 8; ++r) {
      const int mg = s0 + r + 8 * half, jg = j0 + n;
      const bool valid = jg <= mg;
      const float s  = c[r] * SCALE;
      const float nm = fmaxf(m1[r], redmax16(valid ? s : NEG_INF));
      const float p  = valid ? __expf(s - nm) : 0.f;
      d1[r] = d1[r] * __expf(m1[r] - nm) + redsum16(p);
      m1[r] = nm;
    }
  }
#pragma unroll
  for (int r = 0; r < 8; ++r) d1[r] = fmaxf(d1[r], 1e-30f);

  // ---- phase 2: cumsum + decay + online softmax #2 + p.V ----
  float m2[8], l2[8], cum[8], ti[8], alpha[8], pmat[8];
  v8f Oacc[4];
  {
    v8f z = {};
#pragma unroll
    for (int nt = 0; nt < 4; ++nt) Oacc[nt] = z;
  }
#pragma unroll
  for (int r = 0; r < 8; ++r) {
    m2[r] = NEG_INF; l2[r] = 0.f; cum[r] = 0.f;
    ti[r] = (float)tstamp[(size_t)b * Sn + s0 + r + 8 * half];
  }

  for (int jt = 0; jt <= it; ++jt) {
    const int j0 = jt * 16;
    if (jt < it)   // prefetch next K tile (global_prefetch_b8)
      __builtin_prefetch((const void*)(Kb + (base + j0 + 16 + n) * Dn), 0, 1);

    v8f c = {};
#pragma unroll
    for (int kt = 0; kt < 2; ++kt) {
      const v16bf bk =
          *(const v16bf*)(Kb + (base + j0 + n) * Dn + kt * 32 + half * 16);
      c = __builtin_amdgcn_wmma_f32_16x16x32_bf16(false, a[kt], false, bk,
                                                  (short)0, c, false, false);
    }
    const float tj = (float)tstamp[(size_t)b * Sn + j0 + n];

#pragma unroll
    for (int r = 0; r < 8; ++r) {
      const int mg = s0 + r + 8 * half, jg = j0 + n;
      const bool valid = jg <= mg;
      const float s = c[r] * SCALE;

      // stage-1 probability + lane-group inclusive prefix sum (cumsum)
      float pre = valid ? __expf(s - m1[r]) / d1[r] : 0.f;
#pragma unroll
      for (int off = 1; off < 16; off <<= 1) {
        float u = __shfl_up(pre, off, 16);
        if ((l & 15) >= off) pre += u;
      }
      const float tot  = __shfl(pre, 15, 16);        // tile total (carry)
      const float cnew = cum[r] + pre;

      // time-decay reweighted score
      const float dfac = (1.f - cnew) * (ti[r] - tj);
      const float sarr = valid ? s * __expf(-th2 * dfac) : NEG_INF;

      // online softmax #2
      const float nm = fmaxf(m2[r], redmax16(sarr));
      const float p  = valid ? __expf(sarr - nm) : 0.f;
      alpha[r] = __expf(m2[r] - nm);
      l2[r] = l2[r] * alpha[r] + redsum16(p);
      m2[r] = nm;
      cum[r] += tot;
      pmat[r] = p;
    }

    // transpose probabilities C-layout -> A-layout through LDS
    __syncthreads();
#pragma unroll
    for (int r = 0; r < 8; ++r) {
      union { bf16_t b; uint16_t u; } cv; cv.b = f2bf(pmat[r]);
      ldsP[(r + 8 * half) * 17 + n] = cv.u;
    }
    __syncthreads();
    v16bf pa;
#pragma unroll
    for (int i = 0; i < 16; ++i) {
      const int k = a_k(i, half);              // 0..31; only K<16 carries data
      pa[i] = (k < 16) ? bits2bf(ldsP[m * 17 + k]) : bits2bf((uint16_t)0);
    }

    // rescale accumulators and p.V (upper K half zeroed; V is [e][s] major)
#pragma unroll
    for (int nt = 0; nt < 4; ++nt) {
#pragma unroll
      for (int r = 0; r < 8; ++r) Oacc[nt][r] *= alpha[r];
      v16bf bv = {};
      if (half == 0)
        bv = *(const v16bf*)(Vt + ((size_t)bh * Dn + nt * 16 + n) * Sn + j0);
      Oacc[nt] = __builtin_amdgcn_wmma_f32_16x16x32_bf16(false, pa, false, bv,
                                                         (short)0, Oacc[nt],
                                                         false, false);
    }
  }

  // normalize and store per-head context (bf16) for the final projection
#pragma unroll
  for (int nt = 0; nt < 4; ++nt)
#pragma unroll
    for (int r = 0; r < 8; ++r) {
      const float o = Oacc[nt][r] / fmaxf(l2[r], 1e-30f);
      Ob[(base + s0 + r + 8 * half) * Dn + nt * 16 + n] = f2bf(o);
    }
}

// ---------------------------------------------------------------------------
// Kernel 3: output projection  out[b,s,:] = concat_h(O[b,h,s,:]) @ wO (512x64)
// wO pre-transposed to [64][512] bf16 so B-fragments are contiguous.
// ---------------------------------------------------------------------------
__global__ __launch_bounds__(32)
void enc_proj_kernel(const bf16_t* __restrict__ Ob,
                     const bf16_t* __restrict__ wOt,
                     float* __restrict__ out) {
  const int l  = threadIdx.x;
  const int b  = blockIdx.x;
  const int s0 = blockIdx.y * 16;
  const int m  = l & 15, half = l >> 4;
  const int n  = m;

#pragma unroll
  for (int nt = 0; nt < 4; ++nt) {
    v8f c = {};
#pragma unroll
    for (int kt = 0; kt < 16; ++kt) {      // K = 512 in 16 steps of 32
      // A-frag: two contiguous 8-element runs inside one head's D block
      const int k0 = kt * 32 + half * 16;
      const int k1 = kt * 32 + half * 16 + 8;
      const v8bf lo = *(const v8bf*)(Ob + (((size_t)b * Hn + (k0 >> 6)) * Sn + s0 + m) * Dn + (k0 & 63));
      const v8bf hi = *(const v8bf*)(Ob + (((size_t)b * Hn + (k1 >> 6)) * Sn + s0 + m) * Dn + (k1 & 63));
      const v16bf aa = __builtin_shufflevector(lo, hi, 0, 1, 2, 3, 4, 5, 6, 7,
                                               8, 9, 10, 11, 12, 13, 14, 15);
      const v16bf bb =
          *(const v16bf*)(wOt + (size_t)(nt * 16 + n) * (Hn * Dn) + kt * 32 + half * 16);
      c = __builtin_amdgcn_wmma_f32_16x16x32_bf16(false, aa, false, bb,
                                                  (short)0, c, false, false);
    }
#pragma unroll
    for (int r = 0; r < 8; ++r)
      out[((size_t)b * Sn + s0 + r + 8 * half) * Dn + nt * 16 + n] = c[r];
  }
}

// ---------------------------------------------------------------------------
extern "C" void kernel_launch(void* const* d_in, const int* in_sizes, int n_in,
                              void* d_out, int out_size, void* d_ws, size_t ws_size,
                              hipStream_t stream) {
  const float* X     = (const float*)d_in[0];
  const int*   ts    = (const int*)d_in[1];
  const float* wQ    = (const float*)d_in[2];
  const float* wK    = (const float*)d_in[3];
  const float* wV    = (const float*)d_in[4];
  const float* wO    = (const float*)d_in[5];
  const float* theta = (const float*)d_in[6];
  float*       out   = (float*)d_out;

  const size_t nX    = (size_t)Bn * Sn * Dn;        //   262,144
  const size_t nW    = (size_t)Hn * Dn * Dn;        //    32,768 (also = wO size)
  const size_t nBHSD = (size_t)Bn * Hn * Sn * Dn;   // 2,097,152

  bf16_t* Xb  = (bf16_t*)d_ws;
  bf16_t* wQt = Xb + nX;
  bf16_t* wKt = wQt + nW;
  bf16_t* wVt = wKt + nW;
  bf16_t* wOt = wVt + nW;
  bf16_t* Qb  = wOt + nW;
  bf16_t* Kb  = Qb + nBHSD;
  bf16_t* Vt  = Kb + nBHSD;
  bf16_t* Ob  = Vt + nBHSD;

  dim3 blk(32);
  // prepass: convert X, transpose-convert weights
  enc_cvt_kernel<<<dim3((int)(nX / 256)), dim3(256), 0, stream>>>(X, Xb, (int)nX);
  enc_trans_kernel<<<dim3(Hn, Dn * Dn / 256), dim3(256), 0, stream>>>(wQ, wQt, Dn, Dn);
  enc_trans_kernel<<<dim3(Hn, Dn * Dn / 256), dim3(256), 0, stream>>>(wK, wKt, Dn, Dn);
  enc_trans_kernel<<<dim3(Hn, Dn * Dn / 256), dim3(256), 0, stream>>>(wV, wVt, Dn, Dn);
  enc_trans_kernel<<<dim3(1, Hn * Dn * Dn / 256), dim3(256), 0, stream>>>(wO, wOt, Hn * Dn, Dn);

  dim3 gBH(Bn * Hn, Sn / 16);
  enc_qkv_kernel<<<gBH, blk, 0, stream>>>(Xb, wQt, wKt, wVt, Qb, Kb, Vt);
  enc_attn_kernel<<<gBH, blk, 0, stream>>>(Qb, Kb, Vt, ts, theta, Ob);
  enc_proj_kernel<<<dim3(Bn, Sn / 16), blk, 0, stream>>>(Ob, wOt, out);
}
